// surface_816043786335
// MI455X (gfx1250) — matrix operand order repaired and checked by reading, not compiled, not run on hardware
//
#include <hip/hip_runtime.h>
#include <hip/hip_bf16.h>
#include <math.h>

typedef __bf16 bf16;
typedef __attribute__((ext_vector_type(16))) bf16  v16bf;
typedef __attribute__((ext_vector_type(8)))  float v8f;
typedef __attribute__((ext_vector_type(4)))  unsigned int u32x4;
typedef __attribute__((ext_vector_type(8)))  int i32x8;
typedef __attribute__((ext_vector_type(4)))  int i32x4;

#define HSZ 512
#define B_  16
#define S_  50
#define T_  64
#define VSZ 50000
#define G4  2048   // 4*HSZ

union AFrag { v16bf v; unsigned int u[8]; };

__device__ inline float dot4(const float4 a, const float4 b) {
  return fmaf(a.x, b.x, fmaf(a.y, b.y, fmaf(a.z, b.z, a.w * b.w)));
}
__device__ inline float sigmoidf_(float x) { return 1.f / (1.f + expf(-x)); }

// ---- Tensor Data Mover: 2-D contiguous fp32 tile (rows x cols) global -> LDS ----
// D# per CDNA5 ISA: group0 = {count, lds_addr,
// global_addr[56:0], type=2}; group1 = {wg_mask=0, data_size=4B, dims/strides}.
__device__ inline void tdm_load_2d_f32(unsigned lds_off, const void* gptr,
                                       unsigned rows, unsigned cols) {
  unsigned long long ga = (unsigned long long)gptr;
  u32x4 g0;
  g0[0] = 1u;                                        // count=1, user mode, no gather
  g0[1] = lds_off;                                   // lds_addr (bytes)
  g0[2] = (unsigned)(ga & 0xffffffffu);              // global_addr[31:0]
  g0[3] = (unsigned)((ga >> 32) & 0x1ffffffu)        // global_addr[56:32]
          | (2u << 30);                              // type = 2 ("image")
  i32x8 g1;
  g1[0] = (int)(2u << 16);                           // wg_mask=0, data_size=2 (4 bytes)
  g1[1] = (int)(cols << 16);                         // tensor_dim0[15:0] @ [31:16]
  g1[2] = (int)(rows << 16);                         // tensor_dim0 hi=0 | tensor_dim1 lo
  g1[3] = (int)(cols << 16);                         // tensor_dim1 hi=0 | tile_dim0
  g1[4] = (int)rows;                                 // tile_dim1 | tile_dim2=0
  g1[5] = (int)cols;                                 // tensor_dim0_stride[31:0]
  g1[6] = 0;                                         // stride0 hi | stride1 lo
  g1[7] = 0;                                         // stride1 hi
  i32x4 z4 = {0, 0, 0, 0};
#if __clang_major__ >= 23
  i32x8 z8 = {0, 0, 0, 0, 0, 0, 0, 0};
  __builtin_amdgcn_tensor_load_to_lds(g0, g1, z4, z4, z8, 0);
#else
  __builtin_amdgcn_tensor_load_to_lds(g0, g1, z4, z4, 0);
#endif
}

// ---------------- setup / conversion kernels ----------------

__global__ void k_convert_wgen(const float* __restrict__ src, bf16* __restrict__ dst, int n) {
  int i = blockIdx.x * blockDim.x + threadIdx.x;
  int stride = gridDim.x * blockDim.x;
  for (; i < n; i += stride) dst[i] = (bf16)src[i];
}

__global__ void k_convert_wemb(const float* __restrict__ Wih0, bf16* __restrict__ dst) {
  int i = blockIdx.x * blockDim.x + threadIdx.x;   // 0 .. 2048*512-1
  int row = i >> 9, col = i & 511;
  dst[i] = (bf16)Wih0[(size_t)row * 1024 + col];
}

__global__ void k_setup(const float* __restrict__ h0in, const float* __restrict__ c0in,
                        const float* __restrict__ bih0, const float* __restrict__ bhh0,
                        const float* __restrict__ bih1, const float* __restrict__ bhh1,
                        float* h0b0, float* h1b0, float* c0s, float* c1s,
                        float* op, float* bsum0, float* bsum1) {
  int i = blockIdx.x * blockDim.x + threadIdx.x;   // 0..8191
  if (i < B_ * HSZ) {
    h0b0[i] = h0in[i];
    h1b0[i] = h0in[B_ * HSZ + i];
    c0s[i]  = c0in[i];
    c1s[i]  = c0in[B_ * HSZ + i];
    op[i]   = 0.f;
  }
  if (i < G4) { bsum0[i] = bih0[i] + bhh0[i]; bsum1[i] = bih1[i] + bhh1[i]; }
}

__global__ void k_gather_emb(const int* __restrict__ out_tok, const float* __restrict__ emb,
                             bf16* __restrict__ xemb) {
  int m = blockIdx.x;                  // 0..1023 ; m = b*T + t
  int b = m >> 6, tq = m & 63;
  int tok = (tq == 0) ? 3 : out_tok[b * T_ + tq - 1];
  const float* src = emb + (size_t)tok * HSZ;
  bf16* dst = xemb + (size_t)m * HSZ;
  for (int k = threadIdx.x; k < HSZ; k += blockDim.x) dst[k] = (bf16)src[k];
}

// ---------------- WMMA GEMM: C[m,n] = sum_k A[m,k]*W[n,k] + bias[n] ----------------
// 64-row M block staged in padded-stride LDS (stride 260 dwords: 260%64==4 ->
// 16 lanes hit 16 disjoint 4-bank groups per ds_load_b128, and every frag
// address is base + immediate offset). Each wave holds 4 accumulators so every
// B-frag feeds 4 independent WMMAs. N/LDC are compile-time for immediate-offset
// stores.
template <int N, int LDC>
__global__ void k_gemm_bf16(const bf16* __restrict__ A, const bf16* __restrict__ W,
                            const float* __restrict__ bias, float* __restrict__ C) {
  __shared__ unsigned int ldsA[64 * 260];          // 66,560 B (<< 320 KB WGP LDS)
  const int tid = threadIdx.x;
  const int m0 = blockIdx.x * 64;

  const unsigned int* A32 = (const unsigned int*)(A + (size_t)m0 * HSZ);
  #pragma unroll 4
  for (int r = 0; r < 64; ++r)
    ldsA[r * 260 + tid] = A32[(r << 8) + tid];
  __syncthreads();

  const int wave = tid >> 5, lane = tid & 31;
  const int half = lane >> 4, lm = lane & 15;
  const int n0 = (blockIdx.y * 8 + wave) * 16;
  if (n0 >= N) return;                             // whole-wave exit: EXEC all-1s for WMMA

  v8f accs[4] = {};
  // B frag: lane n = lm, contiguous K run starting at k0 + half*16 bf16
  const unsigned int* Wrow = (const unsigned int*)(W + (size_t)(n0 + lm) * HSZ) + half * 8;
  // A frag: row lm (+16 per tile), contiguous K run starting at k0 + half*16 bf16
  const unsigned int* Abase = &ldsA[lm * 260 + half * 8];

  #pragma unroll
  for (int k32 = 0; k32 < 256; k32 += 16) {        // K step = 32 bf16 = 16 dwords
    AFrag bfr;
    #pragma unroll
    for (int i = 0; i < 8; ++i) bfr.u[i] = Wrow[k32 + i];
    AFrag af[4];
    #pragma unroll
    for (int mt = 0; mt < 4; ++mt) {
      const unsigned int* a0 = Abase + mt * (16 * 260) + k32;
      #pragma unroll
      for (int i = 0; i < 4; ++i) { af[mt].u[i] = a0[i]; af[mt].u[4 + i] = a0[4 + i]; }
    }
    #pragma unroll
    for (int mt = 0; mt < 4; ++mt)
      accs[mt] = __builtin_amdgcn_wmma_f32_16x16x32_bf16(false, af[mt].v, false, bfr.v,
                                                         (short)0, accs[mt], false, false);
  }
  const float bv = bias[n0 + lm];
  float* cp = C + (size_t)(m0 + half * 8) * LDC + (n0 + lm);
  #pragma unroll
  for (int mt = 0; mt < 4; ++mt)
    #pragma unroll
    for (int r = 0; r < 8; ++r)                    // VGPR r: lanes<16 M=r, lanes>=16 M=8+r
      cp[(size_t)(mt * 16 + r) * LDC] = accs[mt][r] + bv;
}

// ---------------- recurrent step kernels (L2-bandwidth bound -> fp32 VALU) -------------

__global__ void k_lstm0(const float* __restrict__ Gemb, const float* __restrict__ Wih0,
                        const float* __restrict__ Whh0,
                        const float* __restrict__ op, const float* __restrict__ h_in,
                        float* __restrict__ h_out, float* __restrict__ c_st, int t) {
  __shared__ float glds[4][4][16];
  __shared__ __align__(16) float xs[B_ * HSZ];     // op staged by TDM (32 KB)
  const int tid = threadIdx.x;
  if (tid < 32) {                                  // wave 0 issues one TDM descriptor
    tdm_load_2d_f32((unsigned)(unsigned long long)(void*)xs, op, B_, HSZ);
    __builtin_amdgcn_s_wait_tensorcnt(0);
  }
  __syncthreads();
  const int gate = tid & 3, b = (tid >> 2) & 15, ul = tid >> 6;
  const int u = blockIdx.x * 4 + ul;
  const int j = gate * HSZ + u;                    // PyTorch gate order i,f,g,o
  float acc = Gemb[(size_t)(b * T_ + t) * G4 + j];
  const float4* xo = (const float4*)(xs + b * HSZ);
  const float4* w1 = (const float4*)(Wih0 + (size_t)j * (2 * HSZ) + HSZ);
  const float4* xh = (const float4*)(h_in + b * HSZ);
  const float4* w2 = (const float4*)(Whh0 + (size_t)j * HSZ);
  for (int k = 0; k < HSZ / 4; ++k)
    acc += dot4(xo[k], w1[k]) + dot4(xh[k], w2[k]);
  glds[gate][ul][b] = acc;
  __syncthreads();
  if (tid < 64) {
    int ul2 = tid >> 4, b2 = tid & 15;
    int u2 = blockIdx.x * 4 + ul2;
    float gi = glds[0][ul2][b2], gf = glds[1][ul2][b2];
    float gg = glds[2][ul2][b2], go = glds[3][ul2][b2];
    int idx = b2 * HSZ + u2;
    float c = sigmoidf_(gf) * c_st[idx] + sigmoidf_(gi) * tanhf(gg);
    c_st[idx] = c;
    h_out[idx] = sigmoidf_(go) * tanhf(c);
  }
}

__global__ void k_lstm1(const float* __restrict__ bsum1, const float* __restrict__ Wih1,
                        const float* __restrict__ Whh1,
                        const float* __restrict__ x_in, const float* __restrict__ h_in,
                        float* __restrict__ h_out, float* __restrict__ c_st) {
  __shared__ float glds[4][4][16];
  __shared__ __align__(16) float xs[B_ * HSZ];     // h0-new staged by TDM (32 KB)
  const int tid = threadIdx.x;
  if (tid < 32) {
    tdm_load_2d_f32((unsigned)(unsigned long long)(void*)xs, x_in, B_, HSZ);
    __builtin_amdgcn_s_wait_tensorcnt(0);
  }
  __syncthreads();
  const int gate = tid & 3, b = (tid >> 2) & 15, ul = tid >> 6;
  const int u = blockIdx.x * 4 + ul;
  const int j = gate * HSZ + u;
  float acc = bsum1[j];
  const float4* xo = (const float4*)(xs + b * HSZ);
  const float4* w1 = (const float4*)(Wih1 + (size_t)j * HSZ);
  const float4* xh = (const float4*)(h_in + b * HSZ);
  const float4* w2 = (const float4*)(Whh1 + (size_t)j * HSZ);
  for (int k = 0; k < HSZ / 4; ++k)
    acc += dot4(xo[k], w1[k]) + dot4(xh[k], w2[k]);
  glds[gate][ul][b] = acc;
  __syncthreads();
  if (tid < 64) {
    int ul2 = tid >> 4, b2 = tid & 15;
    int u2 = blockIdx.x * 4 + ul2;
    float gi = glds[0][ul2][b2], gf = glds[1][ul2][b2];
    float gg = glds[2][ul2][b2], go = glds[3][ul2][b2];
    int idx = b2 * HSZ + u2;
    float c = sigmoidf_(gf) * c_st[idx] + sigmoidf_(gi) * tanhf(gg);
    c_st[idx] = c;
    h_out[idx] = sigmoidf_(go) * tanhf(c);
  }
}

// attention + output projection; one WG per batch row
__global__ void k_attn(const float* __restrict__ h1, const float* __restrict__ venc,
                       const unsigned char* __restrict__ mask,
                       const float* __restrict__ Wv, const float* __restrict__ Wout,
                       float* __restrict__ op, bf16* __restrict__ opn_bf, int t) {
  __shared__ float sh[512], svq[512], svcc[512], sw[64];
  const int b = blockIdx.x, tid = threadIdx.x;
  for (int u = tid; u < HSZ; u += 256) sh[u] = h1[b * HSZ + u];
  __syncthreads();
  for (int u = tid; u < HSZ; u += 256) {           // vq = h1 @ Wv^T
    const float4* w = (const float4*)(Wv + (size_t)u * HSZ);
    float a = 0.f;
    for (int k = 0; k < HSZ / 4; ++k) {
      float4 wv = w[k];
      a += wv.x * sh[4*k] + wv.y * sh[4*k+1] + wv.z * sh[4*k+2] + wv.w * sh[4*k+3];
    }
    svq[u] = a;
  }
  __syncthreads();
  if (tid < 64) {                                  // scores + mask
    float sc = -__builtin_inff();
    if (tid < S_) {
      const float4* v = (const float4*)(venc + (size_t)(b * S_ + tid) * HSZ);
      float a = 0.f;
      for (int k = 0; k < HSZ / 4; ++k) {
        float4 vv = v[k];
        a += vv.x * svq[4*k] + vv.y * svq[4*k+1] + vv.z * svq[4*k+2] + vv.w * svq[4*k+3];
      }
      sc = mask[b * S_ + tid] ? -__builtin_inff() : a;
    }
    sw[tid] = sc;
  }
  __syncthreads();
  if (tid == 0) {                                  // softmax over S=50 (tiny)
    float mx = -__builtin_inff();
    for (int s = 0; s < S_; ++s) mx = fmaxf(mx, sw[s]);
    float sum = 0.f;
    for (int s = 0; s < S_; ++s) { float e = expf(sw[s] - mx); sw[s] = e; sum += e; }
    float inv = 1.f / sum;
    for (int s = 0; s < S_; ++s) sw[s] *= inv;
  }
  __syncthreads();
  for (int u = tid; u < HSZ; u += 256) {           // vcc = w @ venc
    float a = 0.f;
    for (int s = 0; s < S_; ++s) a += sw[s] * venc[(size_t)(b * S_ + s) * HSZ + u];
    svcc[u] = a;
  }
  __syncthreads();
  for (int u = tid; u < HSZ; u += 256) {           // opn = tanh([h1,vcc] @ Wout^T)
    const float4* w = (const float4*)(Wout + (size_t)u * (2 * HSZ));
    float a = 0.f;
    for (int k = 0; k < HSZ / 4; ++k) {
      float4 wv = w[k];
      a += wv.x * sh[4*k] + wv.y * sh[4*k+1] + wv.z * sh[4*k+2] + wv.w * sh[4*k+3];
    }
    const float4* w2 = w + HSZ / 4;
    for (int k = 0; k < HSZ / 4; ++k) {
      float4 wv = w2[k];
      a += wv.x * svcc[4*k] + wv.y * svcc[4*k+1] + wv.z * svcc[4*k+2] + wv.w * svcc[4*k+3];
    }
    float r = tanhf(a);
    op[b * HSZ + u] = r;
    opn_bf[(size_t)(b * T_ + t) * HSZ + u] = (bf16)r;
  }
}

// ---------------- host ----------------

extern "C" void kernel_launch(void* const* d_in, const int* in_sizes, int n_in,
                              void* d_out, int out_size, void* d_ws, size_t ws_size,
                              hipStream_t stream) {
  const float* venc  = (const float*)d_in[0];
  const float* h0in  = (const float*)d_in[1];
  const float* c0in  = (const float*)d_in[2];
  const unsigned char* mask = (const unsigned char*)d_in[3];
  const int*   outtok = (const int*)d_in[4];
  const float* emb   = (const float*)d_in[5];
  const float* Wih0  = (const float*)d_in[6];
  const float* Whh0  = (const float*)d_in[7];
  const float* bih0  = (const float*)d_in[8];
  const float* bhh0  = (const float*)d_in[9];
  const float* Wih1  = (const float*)d_in[10];
  const float* Whh1  = (const float*)d_in[11];
  const float* bih1  = (const float*)d_in[12];
  const float* bhh1  = (const float*)d_in[13];
  const float* Wv    = (const float*)d_in[14];
  const float* Wout  = (const float*)d_in[15];
  const float* Wgen  = (const float*)d_in[16];
  const float* bgen  = (const float*)d_in[17];
  float* logits = (float*)d_out;

  char* p = (char*)d_ws;
  auto take = [&](size_t bytes) { char* r = p; p += (bytes + 255) & ~(size_t)255; return r; };
  bf16*  wgen_bf = (bf16*)take((size_t)VSZ * HSZ * 2);   // 51.2 MB (L2-resident)
  bf16*  wemb_bf = (bf16*)take((size_t)G4 * HSZ * 2);
  bf16*  xemb    = (bf16*)take((size_t)B_ * T_ * HSZ * 2);
  bf16*  opn_bf  = (bf16*)take((size_t)B_ * T_ * HSZ * 2);
  float* Gemb    = (float*)take((size_t)B_ * T_ * G4 * 4);
  float* bsum0   = (float*)take(G4 * 4);
  float* bsum1   = (float*)take(G4 * 4);
  float* h0b[2]  = {(float*)take(B_ * HSZ * 4), (float*)take(B_ * HSZ * 4)};
  float* h1b[2]  = {(float*)take(B_ * HSZ * 4), (float*)take(B_ * HSZ * 4)};
  float* c0s     = (float*)take(B_ * HSZ * 4);
  float* c1s     = (float*)take(B_ * HSZ * 4);
  float* op      = (float*)take(B_ * HSZ * 4);

  k_convert_wgen<<<4096, 256, 0, stream>>>(Wgen, wgen_bf, VSZ * HSZ);
  k_convert_wemb<<<(G4 * HSZ) / 256, 256, 0, stream>>>(Wih0, wemb_bf);
  k_setup<<<32, 256, 0, stream>>>(h0in, c0in, bih0, bhh0, bih1, bhh1,
                                  h0b[0], h1b[0], c0s, c1s, op, bsum0, bsum1);
  k_gather_emb<<<B_ * T_, 256, 0, stream>>>(outtok, emb, xemb);

  // Gemb = X_emb @ W_ih0[:, :512]^T + (b_ih0 + b_hh0)   (WMMA, M=1024 N=2048 K=512)
  k_gemm_bf16<G4, G4><<<dim3(16, 16), 256, 0, stream>>>(xemb, wemb_bf, bsum0, Gemb);

  for (int t = 0; t < T_; ++t) {
    k_lstm0<<<128, 256, 0, stream>>>(Gemb, Wih0, Whh0, op, h0b[t & 1], h0b[(t + 1) & 1], c0s, t);
    k_lstm1<<<128, 256, 0, stream>>>(bsum1, Wih1, Whh1, h0b[(t + 1) & 1], h1b[t & 1],
                                     h1b[(t + 1) & 1], c1s);
    k_attn<<<B_, 256, 0, stream>>>(h1b[(t + 1) & 1], venc, mask, Wv, Wout, op, opn_bf, t);
  }

  // logits = opn @ Wgen^T + bgen   (WMMA, M=1024 N=50000 K=512; Wgen bf16 lives in L2)
  k_gemm_bf16<VSZ, VSZ><<<dim3(16, (3125 + 7) / 8), 256, 0, stream>>>(opn_bf, wgen_bf,
                                                                      bgen, logits);
}
